// SelfAttention_2207613190566
// MI455X (gfx1250) — matrix operand order repaired and checked by reading, not compiled, not run on hardware
//
#include <hip/hip_runtime.h>
#include <hip/hip_bf16.h>

typedef __attribute__((ext_vector_type(16))) _Float16 v16h;
typedef __attribute__((ext_vector_type(8)))  float    v8f;

#define WMMA_F32_F16(a,b,c) \
  __builtin_amdgcn_wmma_f32_16x16x32_f16(false,(a),false,(b),(short)0,(c),false,false)

constexpr int Bn = 4, Cn = 2048, En = 1024, Hn = 16, HDn = 64;
constexpr float QSCALE = 0.125f;   // 1/sqrt(64), folded into Q at store time

constexpr int LDA  = 40;  // A tile row stride (halves): 80B rows, 16B aligned
constexpr int LDBT = 40;  // transposed B tile row stride (halves)
constexpr int LKV  = 72;  // Ks row stride (key-major, 64 hd + pad), 144B rows
constexpr int LVT  = 40;  // Vt row stride (hd-major, 32 keys + pad)
constexpr int LP   = 40;  // P scratch row stride

union F16x8 { uint4 u; _Float16 h[8]; };

// Async 16B global->LDS copy (GLOBAL_LOAD_ASYNC_TO_LDS_B128, ASYNCcnt).
// LDS operand is the 32-bit LDS offset (generic LDS addr truncated, ISA 10.2).
__device__ inline void async_g2l_b128(void* lds, const void* g) {
  unsigned l = (unsigned)(unsigned long long)lds;
  unsigned long long a = (unsigned long long)g;
  asm volatile("global_load_async_to_lds_b128 %0, %1, off"
               :: "v"(l), "v"(a) : "memory");
}
__device__ inline void wait_async0() {
  asm volatile("s_wait_asynccnt 0" ::: "memory");
}

// A fragment (16x32 f16, ISA 7.12.2): lane(g,m) -> halves K{8g..8g+7} then
// K{16+8g..16+8g+7} of row m. Two contiguous 16B chunks.
__device__ inline v16h load_frag_A(const _Float16* row, int g) {
  union { v16h v; uint4 u[2]; } r;
  r.u[0] = *(const uint4*)(row + 8 * g);
  r.u[1] = *(const uint4*)(row + 16 + 8 * g);
  return r.v;
}
// B fragment (32x16 f16) from a K-contiguous row (transposed tile or K-major
// row): lane(g,n) -> halves K{8g..8g+7} then K{16+8g..16+8g+7} of column n. Two 16B chunks.
__device__ inline v16h load_frag_Bk(const _Float16* row, int g) {
  union { v16h v; uint4 u[2]; } r;
  r.u[0] = *(const uint4*)(row + 8 * g);
  r.u[1] = *(const uint4*)(row + 16 + 8 * g);
  return r.v;
}

// ---------------------------------------------------------------------------
// Kernel 1: qkv = x @ W_attn ; scatter to Qh/Kh/Vh in [B,H,C,HD] f16 layout.
// Block tile 128(M) x 64(N), 8 waves in 4Mx2N; wave tile 32x32 (4 WMMAs,
// 8 ds_load_b128 per K-step -> 2 LDS loads per WMMA).
// ---------------------------------------------------------------------------
__global__ __launch_bounds__(256) void qkv_kernel(
    const float* __restrict__ x, const float* __restrict__ Wa,
    _Float16* __restrict__ Qh, _Float16* __restrict__ Kh,
    _Float16* __restrict__ Vh) {
  __shared__ _Float16 As[128 * LDA];  // M x K, row-major
  __shared__ _Float16 Bt[64 * LDBT];  // N x K, transposed
  int t = threadIdx.x;
  int m0 = blockIdx.x * 128, n0 = blockIdx.y * 64;
  int wave = t >> 5, wm = wave >> 1, wn = wave & 1;
  int lane = t & 31, g = lane >> 4, ln = lane & 15;
  v8f acc[2][2] = {};
  int ar = t >> 1, ac = (t & 1) * 16;    // A fill: 128 rows x 32 cols (16/thr)
  int bn = t & 63, bk = (t >> 6) * 8;    // B fill: 64 cols x 32 k (col segs)
  for (int k0 = 0; k0 < En; k0 += 32) {
    __syncthreads();
    {  // A: 4x float4 coalesced global, pack to 16 halves, 2x 16B LDS stores
      const float4* xp = (const float4*)(x + (size_t)(m0 + ar) * En + k0 + ac);
      float4 f0 = xp[0], f1 = xp[1], f2 = xp[2], f3 = xp[3];
      F16x8 p0, p1;
      p0.h[0] = (_Float16)f0.x; p0.h[1] = (_Float16)f0.y;
      p0.h[2] = (_Float16)f0.z; p0.h[3] = (_Float16)f0.w;
      p0.h[4] = (_Float16)f1.x; p0.h[5] = (_Float16)f1.y;
      p0.h[6] = (_Float16)f1.z; p0.h[7] = (_Float16)f1.w;
      p1.h[0] = (_Float16)f2.x; p1.h[1] = (_Float16)f2.y;
      p1.h[2] = (_Float16)f2.z; p1.h[3] = (_Float16)f2.w;
      p1.h[4] = (_Float16)f3.x; p1.h[5] = (_Float16)f3.y;
      p1.h[6] = (_Float16)f3.z; p1.h[7] = (_Float16)f3.w;
      *(uint4*)&As[ar * LDA + ac]     = p0.u;
      *(uint4*)&As[ar * LDA + ac + 8] = p1.u;
    }
    {  // B: 8 k-strided loads (coalesced across 64-thread groups), transpose
      const float* wp = Wa + (size_t)(k0 + bk) * (3 * En) + n0 + bn;
      F16x8 p;
#pragma unroll
      for (int i = 0; i < 8; ++i) p.h[i] = (_Float16)wp[(size_t)i * (3 * En)];
      *(uint4*)&Bt[bn * LDBT + bk] = p.u;
    }
    __syncthreads();
    v16h a0 = load_frag_A (&As[(wm * 32 + ln) * LDA], g);
    v16h a1 = load_frag_A (&As[(wm * 32 + 16 + ln) * LDA], g);
    v16h b0 = load_frag_Bk(&Bt[(wn * 32 + ln) * LDBT], g);
    v16h b1 = load_frag_Bk(&Bt[(wn * 32 + 16 + ln) * LDBT], g);
    acc[0][0] = WMMA_F32_F16(a0, b0, acc[0][0]);
    acc[0][1] = WMMA_F32_F16(a0, b1, acc[0][1]);
    acc[1][0] = WMMA_F32_F16(a1, b0, acc[1][0]);
    acc[1][1] = WMMA_F32_F16(a1, b1, acc[1][1]);
  }
  // epilogue: tiles never straddle E/H/C boundaries (all multiples of 64/128)
  int region = n0 / En;                 // 0=Q 1=K 2=V (block-uniform)
  int ebase = n0 - region * En;
  int bidx = m0 / Cn, cbase = m0 - bidx * Cn;
#pragma unroll
  for (int am = 0; am < 2; ++am)
#pragma unroll
    for (int bs = 0; bs < 2; ++bs) {
      v8f cc = acc[am][bs];
      int e0 = ebase + wn * 32 + bs * 16;
      int hh = e0 >> 6, d = (e0 & 63) + ln;
      size_t obase = (((size_t)bidx * Hn + hh) * Cn) * HDn + d;
#pragma unroll
      for (int r = 0; r < 8; ++r) {
        int cidx = cbase + wm * 32 + am * 16 + r + 8 * g;
        size_t o = obase + (size_t)cidx * HDn;
        float val = cc[r];
        if (region == 0)      Qh[o] = (_Float16)(val * QSCALE);
        else if (region == 1) Kh[o] = (_Float16)val;
        else                  Vh[o] = (_Float16)val;
      }
    }
}

// ---------------------------------------------------------------------------
// Kernel 2: causal flash attention per (b,h). Block = 128 query rows, 8 waves
// (16 q-rows each). K/V streamed through LDS in 32-key blocks; K tile filled
// with async global->LDS b128 copies (ASYNCcnt path).
// ---------------------------------------------------------------------------
__global__ __launch_bounds__(256) void attn_kernel(
    const _Float16* __restrict__ Qh, const _Float16* __restrict__ Kh,
    const _Float16* __restrict__ Vh, _Float16* __restrict__ Yh) {
  __shared__ _Float16 Ks[32 * LKV];     // key x hd (K-major rows for S frags)
  __shared__ _Float16 Vt[64 * LVT];     // hd x key (K-major rows for PV frags)
  __shared__ _Float16 Ps[8 * 16 * LP];  // per-wave P bounce
  int t = threadIdx.x, wave = t >> 5, lane = t & 31, g = lane >> 4, n = lane & 15;
  int bh = blockIdx.y;                  // b*H + h
  int q0 = blockIdx.x * 128 + wave * 16;
  const _Float16* Qb = Qh + (size_t)bh * Cn * HDn;
  const _Float16* Kb = Kh + (size_t)bh * Cn * HDn;
  const _Float16* Vb = Vh + (size_t)bh * Cn * HDn;

  // Q tile as two A fragments (hd 0..31, 32..63): 4x b128 global loads
  v16h aQ0 = load_frag_A(&Qb[(size_t)(q0 + n) * HDn], g);
  v16h aQ1 = load_frag_A(&Qb[(size_t)(q0 + n) * HDn + 32], g);

  float mrow[8], lrow[8];
  v8f acc[4] = {};
#pragma unroll
  for (int r = 0; r < 8; ++r) { mrow[r] = -3.0e38f; lrow[r] = 0.f; }

  int kvkey = t >> 3, kvhd = (t & 7) * 8;   // Ks fill mapping
  int vthd = t & 63, vtk = (t >> 6) * 8;    // Vt fill mapping
  int nkb = (blockIdx.x + 1) * 4;           // causal: keys <= q0+127
  int qend = q0 + 15;
  for (int kb = 0; kb < nkb; ++kb) {
    int kbase = kb * 32;
    __syncthreads();
    {  // K: async b128 global->LDS; V: 8 coalesced loads transposed in regs
      async_g2l_b128(&Ks[kvkey * LKV + kvhd],
                     &Kb[(size_t)(kbase + kvkey) * HDn + kvhd]);
      F16x8 p;
#pragma unroll
      for (int i = 0; i < 8; ++i) p.h[i] = Vb[(size_t)(kbase + vtk + i) * HDn + vthd];
      *(uint4*)&Vt[vthd * LVT + vtk] = p.u;
      if (kb + 1 < nkb) {                  // prefetch next K/V block
        __builtin_prefetch(&Kb[(size_t)(kbase + 32 + kvkey) * HDn + kvhd], 0, 1);
        __builtin_prefetch(&Vb[(size_t)(kbase + 32 + kvkey) * HDn + kvhd], 0, 1);
      }
      wait_async0();                       // async done before barrier signal
    }
    __syncthreads();
    if (kbase <= qend) {                  // wave-uniform predicate
      // S = Q K^T (hd inner): Ks rows are K-contiguous -> direct b128 frags
      v8f S[2];
#pragma unroll
      for (int sub = 0; sub < 2; ++sub) {
        const _Float16* krow = &Ks[(sub * 16 + n) * LKV];
        v8f s = {};
        s = WMMA_F32_F16(aQ0, load_frag_Bk(krow, g), s);
        s = WMMA_F32_F16(aQ1, load_frag_Bk(krow + 32, g), s);
        S[sub] = s;
      }
      // causal mask
#pragma unroll
      for (int sub = 0; sub < 2; ++sub) {
        int key = kbase + sub * 16 + n;
#pragma unroll
        for (int r = 0; r < 8; ++r)
          if (key > q0 + r + 8 * g) S[sub][r] = -3.0e38f;
      }
      // online softmax: row reductions across the 16 lanes of each half-wave
      float fac[8];
#pragma unroll
      for (int r = 0; r < 8; ++r) {
        float v = fmaxf(S[0][r], S[1][r]);
#pragma unroll
        for (int off = 1; off < 16; off <<= 1) v = fmaxf(v, __shfl_xor(v, off, 16));
        float mnew = fmaxf(mrow[r], v);
        float f  = __expf(mrow[r] - mnew);
        float p0 = __expf(S[0][r] - mnew);
        float p1 = __expf(S[1][r] - mnew);
        float ps = p0 + p1;
#pragma unroll
        for (int off = 1; off < 16; off <<= 1) ps += __shfl_xor(ps, off, 16);
        lrow[r] = lrow[r] * f + ps;
        mrow[r] = mnew;
        fac[r] = f;
        S[0][r] = p0; S[1][r] = p1;
      }
#pragma unroll
      for (int tt = 0; tt < 4; ++tt)
#pragma unroll
        for (int r = 0; r < 8; ++r) acc[tt][r] *= fac[r];
      // bounce P through per-wave LDS to reshape D-layout -> A-layout
      _Float16* Pw = &Ps[wave * 16 * LP];
#pragma unroll
      for (int r = 0; r < 8; ++r) {
        Pw[(r + 8 * g) * LP + n]      = (_Float16)S[0][r];
        Pw[(r + 8 * g) * LP + 16 + n] = (_Float16)S[1][r];
      }
      asm volatile("s_wait_dscnt 0" ::: "memory");
      v16h aP = load_frag_A(&Pw[n * LP], g);
#pragma unroll
      for (int tt = 0; tt < 4; ++tt) {     // O += P @ V (4 hd column tiles)
        v16h bv = load_frag_Bk(&Vt[(tt * 16 + n) * LVT], g);
        acc[tt] = WMMA_F32_F16(aP, bv, acc[tt]);
      }
    }
  }
  // normalize and write y in [B,C,E] f16 layout (E = H*HD interleave)
  int b = bh / Hn, h = bh % Hn;
#pragma unroll
  for (int r = 0; r < 8; ++r) {
    float inv = 1.0f / lrow[r];
    int q = q0 + r + 8 * g;
#pragma unroll
    for (int tt = 0; tt < 4; ++tt)
      Yh[((size_t)b * Cn + q) * En + h * HDn + tt * 16 + n] =
          (_Float16)(acc[tt][r] * inv);
  }
}

// ---------------------------------------------------------------------------
// Kernel 3: out = y @ W_proj. Same 128x64 tiling; f16 A tile filled with
// async global->LDS b128 copies (pure copy, no conversion needed).
// ---------------------------------------------------------------------------
__global__ __launch_bounds__(256) void proj_kernel(
    const _Float16* __restrict__ Yh, const float* __restrict__ Wp,
    float* __restrict__ out) {
  __shared__ _Float16 As[128 * LDA];
  __shared__ _Float16 Bt[64 * LDBT];
  int t = threadIdx.x;
  int m0 = blockIdx.x * 128, n0 = blockIdx.y * 64;
  int wave = t >> 5, wm = wave >> 1, wn = wave & 1;
  int lane = t & 31, g = lane >> 4, ln = lane & 15;
  v8f acc[2][2] = {};
  int ar = t >> 1, ac = (t & 1) * 16;
  int bn = t & 63, bk = (t >> 6) * 8;
  for (int k0 = 0; k0 < En; k0 += 32) {
    __syncthreads();
    {  // A: two async 16B global->LDS copies per thread
      const _Float16* yp = &Yh[(size_t)(m0 + ar) * En + k0 + ac];
      async_g2l_b128(&As[ar * LDA + ac],     yp);
      async_g2l_b128(&As[ar * LDA + ac + 8], yp + 8);
    }
    {  // B: convert f32 weights, transposed store
      const float* wp = Wp + (size_t)(k0 + bk) * En + n0 + bn;
      F16x8 p;
#pragma unroll
      for (int i = 0; i < 8; ++i) p.h[i] = (_Float16)wp[(size_t)i * En];
      *(uint4*)&Bt[bn * LDBT + bk] = p.u;
    }
    wait_async0();
    __syncthreads();
    v16h a0 = load_frag_A (&As[(wm * 32 + ln) * LDA], g);
    v16h a1 = load_frag_A (&As[(wm * 32 + 16 + ln) * LDA], g);
    v16h b0 = load_frag_Bk(&Bt[(wn * 32 + ln) * LDBT], g);
    v16h b1 = load_frag_Bk(&Bt[(wn * 32 + 16 + ln) * LDBT], g);
    acc[0][0] = WMMA_F32_F16(a0, b0, acc[0][0]);
    acc[0][1] = WMMA_F32_F16(a0, b1, acc[0][1]);
    acc[1][0] = WMMA_F32_F16(a1, b0, acc[1][0]);
    acc[1][1] = WMMA_F32_F16(a1, b1, acc[1][1]);
  }
#pragma unroll
  for (int am = 0; am < 2; ++am)
#pragma unroll
    for (int bs = 0; bs < 2; ++bs) {
      v8f cc = acc[am][bs];
      int col = n0 + wn * 32 + bs * 16 + ln;
#pragma unroll
      for (int r = 0; r < 8; ++r)
        out[(size_t)(m0 + wm * 32 + am * 16 + r + 8 * g) * En + col] = cc[r];
    }
}

extern "C" void kernel_launch(void* const* d_in, const int* in_sizes, int n_in,
                              void* d_out, int out_size, void* d_ws, size_t ws_size,
                              hipStream_t stream) {
  (void)in_sizes; (void)n_in; (void)out_size; (void)ws_size;
  const float* x  = (const float*)d_in[0];
  const float* Wa = (const float*)d_in[1];
  const float* Wp = (const float*)d_in[2];
  float* out = (float*)d_out;
  // workspace: Qh,Kh,Vh,Yh each B*H*C*HD f16 = 16 MB -> 64 MB total
  size_t elems = (size_t)Bn * Hn * Cn * HDn;
  _Float16* Qh = (_Float16*)d_ws;
  _Float16* Kh = Qh + elems;
  _Float16* Vh = Kh + elems;
  _Float16* Yh = Vh + elems;
  qkv_kernel <<<dim3((Bn * Cn) / 128, (3 * En) / 64), 256, 0, stream>>>(x, Wa, Qh, Kh, Vh);
  attn_kernel<<<dim3(Cn / 128, Bn * Hn),              256, 0, stream>>>(Qh, Kh, Vh, Yh);
  proj_kernel<<<dim3((Bn * Cn) / 128, En / 64),       256, 0, stream>>>(Yh, Wp, out);
}
